// MultiHeadAttention_146028888454
// MI455X (gfx1250) — matrix operand order repaired and checked by reading, not compiled, not run on hardware
//
#include <hip/hip_runtime.h>
#include <stdint.h>

#define D_MODEL   1024
#define NUM_HEADS 16
#define HEAD_DIM  64
#define SEQ       2048
#define BATCH     2
#define MROWS     (BATCH*SEQ)   // 4096
#define KC        32
#define STA       40            // LDS A-tile stride in halves (16B-aligned chunks, pad vs banks)
#define GM        256           // workgroup M tile (8 waves x 32 rows)
#define NITER     (D_MODEL / KC)

typedef __bf16 v16bf __attribute__((ext_vector_type(16)));
typedef float  v8f   __attribute__((ext_vector_type(8)));

union FragBF { v16bf v; uint4 u[2]; };

__device__ __forceinline__ unsigned short f2bf(float f) {
  unsigned int u = __float_as_uint(f);
  u += 0x7FFFu + ((u >> 16) & 1u);          // round-to-nearest-even
  return (unsigned short)(u >> 16);
}

// CDNA5 async DMA: copy 16B from global to LDS, tracked by ASYNCcnt.
__device__ __forceinline__ void async_g2l_b128(const unsigned short* gptr,
                                               const unsigned short* lptr) {
  unsigned long long g = (unsigned long long)(uintptr_t)gptr;
  unsigned int       l = (unsigned int)(uintptr_t)lptr;   // flat LDS addr low 32 = LDS offset
  asm volatile("global_load_async_to_lds_b128 %0, %1, off"
               :: "v"(l), "v"(g) : "memory");
}

// ---------------------------------------------------------------------------
// Stage 0: fp32 -> bf16 streaming cast (one pass; all later traffic is bf16).
// ---------------------------------------------------------------------------
__global__ __launch_bounds__(256)
void cast_bf16_kernel(const float* __restrict__ src, unsigned short* __restrict__ dst, int n8)
{
  int i = blockIdx.x * 256 + threadIdx.x;      // 8 floats per thread
  if (i >= n8) return;
  const float4* s = (const float4*)src + (size_t)i * 2;
  float4 f0 = s[0], f1 = s[1];
  uint4 o;
  o.x = f2bf(f0.x) | ((unsigned)f2bf(f0.y) << 16);
  o.y = f2bf(f0.z) | ((unsigned)f2bf(f0.w) << 16);
  o.z = f2bf(f1.x) | ((unsigned)f2bf(f1.y) << 16);
  o.w = f2bf(f1.z) | ((unsigned)f2bf(f1.w) << 16);
  ((uint4*)dst)[i] = o;
}

// ---------------------------------------------------------------------------
// bf16 GEMM: C = X @ W^T + bias.  256x64 WG tile, 32x64 per wave, K-step 32.
// Double-buffered LDS filled by global_load_async_to_lds_b128.
//   F32OUT=false: bf16 per-head scatter (vlayout 0: [B,H,S,Dh]; 1: [B,H,Dh,S])
//   F32OUT=true : fp32 row-major [4096,1024]
// ---------------------------------------------------------------------------
template<bool F32OUT>
__global__ __launch_bounds__(256)
void gemm_bf16_kernel(const unsigned short* __restrict__ X,   // [4096,1024] bf16
                      const unsigned short* __restrict__ W,   // [1024,1024] bf16 (rows = out cols)
                      const float* __restrict__ bias,
                      void* __restrict__ outp, int vlayout)
{
  __shared__ unsigned short lA[2][GM * STA];   // 2 x 20 KB
  __shared__ unsigned short lB[2][64 * 32];    // 2 x  4 KB

  const int tid  = threadIdx.x;
  const int wave = tid >> 5;
  const int lane = tid & 31;
  const int l16  = lane & 15;
  const int hi   = lane >> 4;

  const int rowBase = blockIdx.x * GM;
  const int colBase = blockIdx.y * 64;

  v8f acc[2][4] = {};

  // Issue one K-tile's async copies into buffer `buf` (5 x b128 per thread).
  auto stage = [&](int k0, int buf) {
    #pragma unroll
    for (int j = 0; j < 4; ++j) {                 // A: 256 rows x 32 halves
      int op = j * 256 + tid;                     // 0..1023
      int r  = op >> 2;
      int kc = (op & 3) << 3;                     // halves: 0,8,16,24
      async_g2l_b128(X + (size_t)(rowBase + r) * D_MODEL + k0 + kc,
                     &lA[buf][r * STA + kc]);
    }
    {                                             // B: 64 rows x 32 halves
      int r  = tid >> 2;
      int kc = (tid & 3) << 3;
      async_g2l_b128(W + (size_t)(colBase + r) * D_MODEL + k0 + kc,
                     &lB[buf][r * 32 + kc]);
    }
  };

  stage(0, 0);
  for (int it = 0; it < NITER; ++it) {
    const int buf = it & 1;
    if (it + 1 < NITER) {
      stage((it + 1) * KC, buf ^ 1);
      asm volatile("s_wait_asynccnt 0x5" ::: "memory");  // batch(it) done; batch(it+1) in flight
    } else {
      asm volatile("s_wait_asynccnt 0x0" ::: "memory");
    }
    __syncthreads();

    const unsigned short* As = &lA[buf][0];
    const unsigned short* Bs = &lB[buf][0];
    FragBF a[2];
    #pragma unroll
    for (int ar = 0; ar < 2; ++ar) {
      const unsigned short* ab = As + (wave * 32 + ar * 16 + l16) * STA + hi * 8;
      a[ar].u[0] = *(const uint4*)(ab);
      a[ar].u[1] = *(const uint4*)(ab + 16);
    }
    #pragma unroll
    for (int t = 0; t < 4; ++t) {
      FragBF b;
      const unsigned short* bp = Bs + (t * 16 + l16) * 32 + hi * 8;
      b.u[0] = *(const uint4*)(bp);
      b.u[1] = *(const uint4*)(bp + 16);
      acc[0][t] = __builtin_amdgcn_wmma_f32_16x16x32_bf16(
                    false, a[0].v, false, b.v, (short)0, acc[0][t], false, false);
      acc[1][t] = __builtin_amdgcn_wmma_f32_16x16x32_bf16(
                    false, a[1].v, false, b.v, (short)0, acc[1][t], false, false);
    }
    __syncthreads();
  }

  #pragma unroll
  for (int ar = 0; ar < 2; ++ar) {
    #pragma unroll
    for (int t = 0; t < 4; ++t) {
      int c = colBase + t * 16 + l16;
      float bv = bias[c];
      #pragma unroll
      for (int e = 0; e < 8; ++e) {
        int m = rowBase + wave * 32 + ar * 16 + hi * 8 + e;
        float val = acc[ar][t][e] + bv;
        if constexpr (F32OUT) {
          ((float*)outp)[(size_t)m * D_MODEL + c] = val;
        } else {
          int bb = m >> 11, s = m & 2047, h = c >> 6, d = c & 63;
          unsigned short* O = (unsigned short*)outp;
          if (!vlayout)
            O[((size_t)(bb * NUM_HEADS + h) * SEQ + s) * HEAD_DIM + d] = f2bf(val);
          else
            O[((size_t)(bb * NUM_HEADS + h) * HEAD_DIM + d) * SEQ + s] = f2bf(val);
        }
      }
    }
  }
}

// ---------------------------------------------------------------------------
// Stage 2: flash attention.  One WG per (b*h, 128 q rows); wave = 16 q rows.
// ---------------------------------------------------------------------------
__global__ __launch_bounds__(256)
void attn_kernel(const unsigned short* __restrict__ Q,   // [B,H,S,Dh] bf16
                 const unsigned short* __restrict__ K,   // [B,H,S,Dh] bf16
                 const unsigned short* __restrict__ V,   // [B,H,Dh,S] bf16
                 unsigned short* __restrict__ Oa)        // [B,S,H*Dh] bf16
{
  __shared__ unsigned short lP[8][16 * 64];  // per-wave P relay (C-layout -> A-layout)

  const int tid  = threadIdx.x;
  const int wave = tid >> 5;
  const int lane = tid & 31;
  const int l16  = lane & 15;
  const int hi   = lane >> 4;

  const int bh    = blockIdx.y;
  const int qbase = blockIdx.x * 128 + wave * 16;
  const size_t qkBase = (size_t)bh * SEQ * HEAD_DIM;
  const size_t vBase  = (size_t)bh * HEAD_DIM * SEQ;

  FragBF qa[2];
  {
    const unsigned short* qp = Q + qkBase + (size_t)(qbase + l16) * HEAD_DIM;
    #pragma unroll
    for (int c = 0; c < 2; ++c) {
      qa[c].u[0] = *(const uint4*)(qp + c * 32 + hi * 8);
      qa[c].u[1] = *(const uint4*)(qp + c * 32 + 16 + hi * 8);
    }
  }

  v8f o[4] = {};
  float mI[8], lI[8];
  #pragma unroll
  for (int e = 0; e < 8; ++e) { mI[e] = -1e30f; lI[e] = 0.f; }

  const float lam = 0.125f * 1.44269504088896340736f;  // 1/sqrt(Dh) * log2(e)

  for (int kb = 0; kb < SEQ; kb += 64) {
    if (kb + 64 < SEQ) {   // prefetch next K/V tiles (global_prefetch_b8)
      __builtin_prefetch(K + qkBase + (size_t)(kb + 64 + lane) * HEAD_DIM, 0, 1);
      __builtin_prefetch(V + vBase + (size_t)(l16) * SEQ + kb + 64 + hi * 32, 0, 1);
    }

    // ---- S = Q K^T  (16 x 64 per wave) ----
    v8f s[4] = {};
    #pragma unroll
    for (int t = 0; t < 4; ++t) {
      const unsigned short* kp = K + qkBase + (size_t)(kb + t * 16 + l16) * HEAD_DIM;
      #pragma unroll
      for (int c = 0; c < 2; ++c) {
        FragBF b;
        b.u[0] = *(const uint4*)(kp + c * 32 + hi * 8);
        b.u[1] = *(const uint4*)(kp + c * 32 + hi * 8 + 16);
        s[t] = __builtin_amdgcn_wmma_f32_16x16x32_bf16(
                 false, qa[c].v, false, b.v, (short)0, s[t], false, false);
      }
    }

    // ---- online softmax in exp2 domain ----
    float mx[8];
    #pragma unroll
    for (int e = 0; e < 8; ++e) {
      float v0 = fmaxf(fmaxf(s[0][e], s[1][e]), fmaxf(s[2][e], s[3][e])) * lam;
      v0 = fmaxf(v0, __shfl_xor(v0, 1, 32));
      v0 = fmaxf(v0, __shfl_xor(v0, 2, 32));
      v0 = fmaxf(v0, __shfl_xor(v0, 4, 32));
      v0 = fmaxf(v0, __shfl_xor(v0, 8, 32));
      mx[e] = v0;
    }
    float alpha[8];
    #pragma unroll
    for (int e = 0; e < 8; ++e) {
      float nm = fmaxf(mI[e], mx[e]);
      alpha[e] = __builtin_amdgcn_exp2f(mI[e] - nm);
      mI[e] = nm;
    }

    unsigned short* pw = &lP[wave][0];
    float psum[8];
    #pragma unroll
    for (int e = 0; e < 8; ++e) psum[e] = 0.f;
    #pragma unroll
    for (int t = 0; t < 4; ++t) {
      #pragma unroll
      for (int e = 0; e < 8; ++e) {
        float p = __builtin_amdgcn_exp2f(s[t][e] * lam - mI[e]);
        psum[e] += p;
        pw[(hi * 8 + e) * 64 + t * 16 + l16] = f2bf(p);
      }
    }
    #pragma unroll
    for (int e = 0; e < 8; ++e) {
      float ps = psum[e];
      ps += __shfl_xor(ps, 1, 32);
      ps += __shfl_xor(ps, 2, 32);
      ps += __shfl_xor(ps, 4, 32);
      ps += __shfl_xor(ps, 8, 32);
      lI[e] = lI[e] * alpha[e] + ps;
      o[0][e] *= alpha[e]; o[1][e] *= alpha[e];
      o[2][e] *= alpha[e]; o[3][e] *= alpha[e];
    }

    // Wave-private LDS region: wave-level DS-counter wait suffices (no WG barrier).
    asm volatile("s_wait_dscnt 0x0" ::: "memory");

    // ---- O += P V ----
    #pragma unroll
    for (int c = 0; c < 2; ++c) {
      FragBF pa;
      const unsigned short* pr = pw + l16 * 64 + c * 32 + hi * 8;
      pa.u[0] = *(const uint4*)(pr);
      pa.u[1] = *(const uint4*)(pr + 16);
      #pragma unroll
      for (int t = 0; t < 4; ++t) {
        FragBF vb;
        const unsigned short* vp = V + vBase + (size_t)(t * 16 + l16) * SEQ
                                     + kb + c * 32 + hi * 8;
        vb.u[0] = *(const uint4*)(vp);
        vb.u[1] = *(const uint4*)(vp + 16);
        o[t] = __builtin_amdgcn_wmma_f32_16x16x32_bf16(
                 false, pa.v, false, vb.v, (short)0, o[t], false, false);
      }
    }
  }

  const int b = bh >> 4, h = bh & 15;
  #pragma unroll
  for (int e = 0; e < 8; ++e) {
    float inv = 1.0f / lI[e];
    int srow = qbase + hi * 8 + e;
    size_t rb = (size_t)(b * SEQ + srow) * D_MODEL + h * HEAD_DIM;
    #pragma unroll
    for (int t = 0; t < 4; ++t)
      Oa[rb + t * 16 + l16] = f2bf(o[t][e] * inv);
  }
}

// ---------------------------------------------------------------------------
extern "C" void kernel_launch(void* const* d_in, const int* in_sizes, int n_in,
                              void* d_out, int out_size, void* d_ws, size_t ws_size,
                              hipStream_t stream)
{
  const float* q  = (const float*)d_in[0];
  const float* k  = (const float*)d_in[1];
  const float* v  = (const float*)d_in[2];
  const float* wq = (const float*)d_in[3];
  const float* bq = (const float*)d_in[4];
  const float* wk = (const float*)d_in[5];
  const float* bk = (const float*)d_in[6];
  const float* wv = (const float*)d_in[7];
  const float* bv = (const float*)d_in[8];
  const float* wo = (const float*)d_in[9];
  const float* bo = (const float*)d_in[10];

  const size_t headElems = (size_t)BATCH * NUM_HEADS * SEQ * HEAD_DIM;  // 4M
  unsigned short* ws_q = (unsigned short*)d_ws;       // 8 MB
  unsigned short* ws_k = ws_q + headElems;            // 8 MB
  unsigned short* ws_v = ws_k + headElems;            // 8 MB
  unsigned short* Xb   = ws_v + headElems;            // 8 MB: bf16 X, later attn out
  unsigned short* Wb   = Xb + headElems;              // 2 MB: bf16 weight (reused)

  dim3 blk(256, 1, 1);
  dim3 ggrid(MROWS / GM, D_MODEL / 64, 1);            // 16 x 16
  const int NX8 = MROWS * D_MODEL / 8;                // 524288
  const int NW8 = D_MODEL * D_MODEL / 8;              // 131072

  // Q projection
  cast_bf16_kernel<<<NX8 / 256, blk, 0, stream>>>(q,  Xb, NX8);
  cast_bf16_kernel<<<NW8 / 256, blk, 0, stream>>>(wq, Wb, NW8);
  gemm_bf16_kernel<false><<<ggrid, blk, 0, stream>>>(Xb, Wb, bq, ws_q, 0);
  // K projection
  cast_bf16_kernel<<<NX8 / 256, blk, 0, stream>>>(k,  Xb, NX8);
  cast_bf16_kernel<<<NW8 / 256, blk, 0, stream>>>(wk, Wb, NW8);
  gemm_bf16_kernel<false><<<ggrid, blk, 0, stream>>>(Xb, Wb, bk, ws_k, 0);
  // V projection (pre-transposed output)
  cast_bf16_kernel<<<NX8 / 256, blk, 0, stream>>>(v,  Xb, NX8);
  cast_bf16_kernel<<<NW8 / 256, blk, 0, stream>>>(wv, Wb, NW8);
  gemm_bf16_kernel<false><<<ggrid, blk, 0, stream>>>(Xb, Wb, bv, ws_v, 1);
  // Attention (writes into Xb, which is free now)
  attn_kernel<<<dim3(SEQ / 128, BATCH * NUM_HEADS, 1), blk, 0, stream>>>(
      ws_q, ws_k, ws_v, Xb);
  // Output projection
  cast_bf16_kernel<<<NW8 / 256, blk, 0, stream>>>(wo, Wb, NW8);
  gemm_bf16_kernel<true><<<ggrid, blk, 0, stream>>>(Xb, Wb, bo, (float*)d_out, 0);
}
